// PointerGen_48627619725545
// MI455X (gfx1250) — matrix operand order repaired and checked by reading, not compiled, not run on hardware
//
#include <hip/hip_runtime.h>
#include <hip/hip_bf16.h>
#include <math.h>

typedef __attribute__((ext_vector_type(16))) __bf16 v16bf;
typedef __attribute__((ext_vector_type(8)))  float  v8f;

#define BB   256
#define SS   400
#define HH   512
#define EMBS 1024
#define H3   1536
#define VV   50257
#define VP   50357   // V + EXTRA

// d_out layout (floats): out[B,VP] | hidden[B,H] | attn_weights[S,B] | sig_hid[B]
#define OFS_HID 12891392L
#define OFS_AW  13022464L
#define OFS_SIG 13124864L

// ---------- helpers ----------
__device__ __forceinline__ __bf16 f2bf(float f) {
  unsigned u = __builtin_bit_cast(unsigned, f);
  u = u + 0x7FFFu + ((u >> 16) & 1u);               // round-to-nearest-even
  unsigned short h = (unsigned short)(u >> 16);
  return __builtin_bit_cast(__bf16, h);
}
__device__ __forceinline__ float gelu_exact(float x) {
  return 0.5f * x * (1.0f + erff(x * 0.70710678118654752f));
}
__device__ __forceinline__ float sigmoidf_(float x) { return 1.0f / (1.0f + __expf(-x)); }

__device__ __forceinline__ v8f wmma_bf16(v16bf a, v16bf b, v8f c) {
  return __builtin_amdgcn_wmma_f32_16x16x32_bf16(false, a, false, b, (short)0, c, false, false);
}
__device__ __forceinline__ v8f vzero8() {
  v8f z;
#pragma unroll
  for (int i = 0; i < 8; ++i) z[i] = 0.0f;
  return z;
}
// Fragment-major operand: frag f = 32 lanes x 16 halves contiguous (1KB).
// Per-lane load = one 32B vector load (2x b128).
__device__ __forceinline__ v16bf load_frag(const __bf16* P, int fragIdx) {
  int l = threadIdx.x & 31;
  return ((const v16bf*)P)[fragIdx * 32 + l];
}

// ---------- swizzle passes: fp32 row-major -> bf16 fragment-major ----------
// A-fragment (16x32): lane l holds row m=l&15; halves j map to
// k = kt*32 + ((l>=16)?8:0) + ((j>>3)<<4) + (j&7)
__global__ void __launch_bounds__(256) k_swz_a(const float* src, __bf16* dst,
                                               int ldw, int kofs, int KT, long total) {
  long e = (long)blockIdx.x * 256 + threadIdx.x;
  if (e >= total) return;
  int j  = (int)(e & 15);
  int l  = (int)((e >> 4) & 31);
  long f = e >> 9;
  int kt = (int)(f % KT);
  int mt = (int)(f / KT);
  int m   = mt * 16 + (l & 15);
  int khi = ((l >> 4) << 3) + ((j >> 3) << 4) + (j & 7);
  dst[e] = f2bf(src[(long)m * ldw + kofs + kt * 32 + khi]);
}
// B-fragment (32x16): lane l holds n=l&15; halves j map to k like A.  src is W[N][K] row-major.
__global__ void __launch_bounds__(256) k_swz_b(const float* src, __bf16* dst,
                                               int ldw, int kofs, int KT, long total) {
  long e = (long)blockIdx.x * 256 + threadIdx.x;
  if (e >= total) return;
  int j  = (int)(e & 15);
  int l  = (int)((e >> 4) & 31);
  long f = e >> 9;
  int kt = (int)(f % KT);
  int nt = (int)(f / KT);
  dst[e] = f2bf(src[(long)(nt * 16 + (l & 15)) * ldw + kofs + kt * 32 + ((l >> 4) << 3) + ((j >> 3) << 4) + (j & 7)]);
}

// ---------- 1. GRU input/hidden GEMMs ----------
__global__ void __launch_bounds__(32) k_gru_gemm(const __bf16* xsw, const __bf16* hpsw,
    const __bf16* Wihsw, const __bf16* Whhsw, const float* bih, const float* bhh,
    float* gi, float* gh) {
  int mt = blockIdx.x, nt = blockIdx.y;
  int l = threadIdx.x & 31;
  int n  = nt * 16 + (l & 15);
  int mb = mt * 16 + ((l >> 4) << 3);

  v8f acc = vzero8();
  for (int kt = 0; kt < 32; ++kt)
    acc = wmma_bf16(load_frag(xsw, mt * 32 + kt), load_frag(Wihsw, nt * 32 + kt), acc);
  float bi = bih[n];
#pragma unroll
  for (int r = 0; r < 8; ++r) gi[(long)(mb + r) * H3 + n] = acc[r] + bi;

  acc = vzero8();
  for (int kt = 0; kt < 16; ++kt)
    acc = wmma_bf16(load_frag(hpsw, mt * 16 + kt), load_frag(Whhsw, nt * 16 + kt), acc);
  float bh = bhh[n];
#pragma unroll
  for (int r = 0; r < 8; ++r) gh[(long)(mb + r) * H3 + n] = acc[r] + bh;
}

// ---------- 2. GRU gates + hidden out + sig_hid ----------
__global__ void __launch_bounds__(128) k_gates(const float* gi, const float* gh,
    const float* hprev, const float* sigW, const float* sigb,
    float* cc, float* dout) {
  int b = blockIdx.x, t = threadIdx.x;
  __shared__ float red[128];
  float p = 0.0f;
  for (int h = t; h < HH; h += 128) {
    float ir = gi[(long)b * H3 + h], iz = gi[(long)b * H3 + 512 + h], in_ = gi[(long)b * H3 + 1024 + h];
    float hr = gh[(long)b * H3 + h], hz = gh[(long)b * H3 + 512 + h], hn = gh[(long)b * H3 + 1024 + h];
    float r = sigmoidf_(ir + hr);
    float z = sigmoidf_(iz + hz);
    float nn = tanhf(in_ + r * hn);
    float hp = hprev[(long)b * HH + h];
    float o = (1.0f - z) * nn + z * hp;
    cc[(long)b * EMBS + h] = o;                 // rnn_output into concat buffer cols [0,512)
    dout[OFS_HID + (long)b * HH + h] = o;       // hidden output
    p += o * sigW[h];
  }
  red[t] = p; __syncthreads();
  for (int s = 64; s > 0; s >>= 1) { if (t < s) red[t] += red[t + s]; __syncthreads(); }
  if (t == 0) dout[OFS_SIG + b] = red[0] + sigb[0];
}

// ---------- 3. hpart = rnn_output @ attn_W[:, :H].T ----------
__global__ void __launch_bounds__(32) k_hpart(const __bf16* rnnsw, const __bf16* aWhsw,
                                              float* hpart) {
  int mt = blockIdx.x, nt = blockIdx.y;
  v8f acc = vzero8();
  for (int kt = 0; kt < 16; ++kt)
    acc = wmma_bf16(load_frag(rnnsw, mt * 16 + kt), load_frag(aWhsw, nt * 16 + kt), acc);
  int l = threadIdx.x & 31;
  int n  = nt * 16 + (l & 15);
  int mb = mt * 16 + ((l >> 4) << 3);
#pragma unroll
  for (int r = 0; r < 8; ++r) hpart[(long)(mb + r) * HH + n] = acc[r];
}

// ---------- 4. energy GEMM + tanh + v-dot -> scores ----------
__global__ void __launch_bounds__(256) k_energy(const float* enc, const __bf16* aWesw,
    const float* attnb, const float* attnv, const float* hpart, float* scores) {
  __shared__ float tAf[16 * HH];                               // 32 KB staged fp32 tile
  __shared__ __attribute__((aligned(32))) __bf16 tA[16 * HH];  // 16 KB fragment-major bf16
  __shared__ float sS[16];
  int t = threadIdx.x;
  int row0 = blockIdx.x * 16;                  // rows are flattened (s*B + b); tile contiguous

  // async DMA the contiguous 32KB A-tile into LDS (8 x b128 per wave)
  const float* gsrc = enc + (long)row0 * HH;
  for (int it = 0; it < 8; ++it) {
    unsigned lofs = (unsigned)(unsigned long long)(&tAf[0]) + (unsigned)((it * 256 + t) * 16);
    const float* ga = gsrc + (it * 256 + t) * 4;
    asm volatile("global_load_async_to_lds_b128 %0, %1, off" :: "v"(lofs), "v"(ga) : "memory");
  }
  asm volatile("s_wait_asynccnt 0x0" ::: "memory");
  __syncthreads();

  // swizzle fp32 tile -> bf16 fragment-major tile
  for (int i = t; i < 16 * HH; i += 256) {
    int j = i & 15, l2 = (i >> 4) & 31, kt = i >> 9;
    int m   = l2 & 15;
    int khi = ((l2 >> 4) << 3) + ((j >> 3) << 4) + (j & 7);
    tA[i] = f2bf(tAf[m * HH + kt * 32 + khi]);
  }
  if (t < 16) sS[t] = 0.0f;
  __syncthreads();

  int w = t >> 5, l = t & 31;
  v8f acc[4];
#pragma unroll
  for (int nt = 0; nt < 4; ++nt) acc[nt] = vzero8();

  const v16bf* Ap = (const v16bf*)tA;
  for (int kt = 0; kt < 16; ++kt) {
    v16bf a = Ap[kt * 32 + l];                 // 2x ds_load_b128, reused 4x
#pragma unroll
    for (int nt = 0; nt < 4; ++nt) {
      int ntg = w * 4 + nt;
      acc[nt] = wmma_bf16(a, load_frag(aWesw, ntg * 16 + kt), acc[nt]);
    }
  }

  int mb = (l >> 4) << 3;
  float sacc[8];
#pragma unroll
  for (int r = 0; r < 8; ++r) sacc[r] = 0.0f;
#pragma unroll
  for (int nt = 0; nt < 4; ++nt) {
    int n = (w * 4 + nt) * 16 + (l & 15);
    float vn = attnv[n], bn = attnb[n];
#pragma unroll
    for (int r = 0; r < 8; ++r) {
      int rowg = row0 + mb + r;
      int b = rowg & 255;
      float e = tanhf(acc[nt][r] + hpart[(long)b * HH + n] + bn);
      sacc[r] += e * vn;
    }
  }
#pragma unroll
  for (int r = 0; r < 8; ++r) atomicAdd(&sS[mb + r], sacc[r]);
  __syncthreads();
  if (t < 16) {
    int rowg = row0 + t;
    int b = rowg & 255, s = rowg >> 8;
    scores[(long)b * SS + s] = sS[t];
  }
}

// ---------- 5. masked softmax over S ----------
__global__ void __launch_bounds__(256) k_softmax(const float* scores, const float* mask,
                                                 float* attnw, float* dout) {
  int b = blockIdx.x, t = threadIdx.x;
  __shared__ float red[256];
  float m = -3.0e38f;
  for (int s = t; s < SS; s += 256) {
    float x = (mask[(long)b * SS + s] > 0.0f) ? scores[(long)b * SS + s] : -1.0e18f;
    m = fmaxf(m, x);
  }
  red[t] = m; __syncthreads();
  for (int s = 128; s > 0; s >>= 1) { if (t < s) red[t] = fmaxf(red[t], red[t + s]); __syncthreads(); }
  m = red[0]; __syncthreads();
  float sum = 0.0f;
  for (int s = t; s < SS; s += 256) {
    float x = (mask[(long)b * SS + s] > 0.0f) ? scores[(long)b * SS + s] : -1.0e18f;
    sum += __expf(x - m);
  }
  red[t] = sum; __syncthreads();
  for (int s = 128; s > 0; s >>= 1) { if (t < s) red[t] += red[t + s]; __syncthreads(); }
  float inv = 1.0f / red[0]; __syncthreads();
  for (int s = t; s < SS; s += 256) {
    float x = (mask[(long)b * SS + s] > 0.0f) ? scores[(long)b * SS + s] : -1.0e18f;
    float wv = __expf(x - m) * inv;
    attnw[(long)b * SS + s] = wv;
    dout[OFS_AW + (long)s * BB + b] = wv;       // attn_weights output [S,B]
  }
}

// ---------- 6. context = sum_s attn_w * enc ----------
__global__ void __launch_bounds__(256) k_context(const float* enc, const float* attnw, float* cc) {
  int b = blockIdx.x;
  int h = blockIdx.y * 256 + threadIdx.x;
  __shared__ float wrow[SS];
  for (int s = threadIdx.x; s < SS; s += 256) wrow[s] = attnw[(long)b * SS + s];
  __syncthreads();
  const float* ep = enc + (long)b * HH + h;
  float acc = 0.0f;
  for (int s = 0; s < SS; ++s) {
    if (s + 8 < SS) __builtin_prefetch(ep + (long)(s + 8) * (BB * HH), 0, 1);
    acc += wrow[s] * ep[(long)s * (BB * HH)];
  }
  cc[(long)b * EMBS + HH + h] = acc;            // concat buffer cols [512,1024)
}

// ---------- 7. pgen gate ----------
__global__ void __launch_bounds__(256) k_pgen(const float* hprev, const float* cc, const float* x,
                                              const float* pW, const float* pb, float* pgen) {
  int b = blockIdx.x, t = threadIdx.x;
  __shared__ float red[256];
  float p = 0.0f;
  for (int i = t; i < 2 * HH + EMBS; i += 256) {
    float v;
    if (i < HH)            v = hprev[(long)b * HH + i];
    else if (i < 2 * HH)   v = cc[(long)b * EMBS + HH + (i - HH)];
    else                   v = x[(long)b * EMBS + (i - 2 * HH)];
    p += v * pW[i];
  }
  red[t] = p; __syncthreads();
  for (int s = 128; s > 0; s >>= 1) { if (t < s) red[t] += red[t + s]; __syncthreads(); }
  if (t == 0) pgen[b] = sigmoidf_(gelu_exact(red[0] + pb[0]));
}

// ---------- 8. concat_out = gelu([rnn;ctx] @ concat_W.T + b) ----------
__global__ void __launch_bounds__(32) k_concat(const __bf16* ccsw, const __bf16* cWsw,
                                               const float* cb, float* co) {
  int mt = blockIdx.x, nt = blockIdx.y;
  v8f acc = vzero8();
  for (int kt = 0; kt < 32; ++kt)
    acc = wmma_bf16(load_frag(ccsw, mt * 32 + kt), load_frag(cWsw, nt * 32 + kt), acc);
  int l = threadIdx.x & 31;
  int n  = nt * 16 + (l & 15);
  int mb = mt * 16 + ((l >> 4) << 3);
  float bias = cb[n];
#pragma unroll
  for (int r = 0; r < 8; ++r) co[(long)(mb + r) * HH + n] = gelu_exact(acc[r] + bias);
}

// ---------- 9. vocab logits into d_out (4 m-tiles per block) ----------
__global__ void __launch_bounds__(32) k_logits(const __bf16* cosw, const float* outW,
                                               const float* outb, float* out) {
  int n0 = blockIdx.y * 16;
  int l = threadIdx.x & 31;
  v8f acc[4];
#pragma unroll
  for (int mt = 0; mt < 4; ++mt) acc[mt] = vzero8();

  for (int kt = 0; kt < 16; ++kt) {
    v16bf bf;
#pragma unroll
    for (int j = 0; j < 16; ++j) {
      int n = n0 + (l & 15); if (n >= VV) n = VV - 1;   // edge-tile clamp; stores guarded
      bf[j] = f2bf(outW[(long)n * HH + kt * 32 + ((l >> 4) << 3) + ((j >> 3) << 4) + (j & 7)]);
    }
#pragma unroll
    for (int mt = 0; mt < 4; ++mt) {
      int mtg = blockIdx.x * 4 + mt;
      acc[mt] = wmma_bf16(load_frag(cosw, mtg * 16 + kt), bf, acc[mt]);
    }
  }
  int n = n0 + (l & 15);
  if (n < VV) {
    float bias = outb[n];
#pragma unroll
    for (int mt = 0; mt < 4; ++mt) {
      int mb = (blockIdx.x * 4 + mt) * 16 + ((l >> 4) << 3);
#pragma unroll
      for (int r = 0; r < 8; ++r) out[(long)(mb + r) * VP + n] = acc[mt][r] + bias;
    }
  }
}

// ---------- 10. per-row max & sum(exp) over V ----------
__global__ void __launch_bounds__(256) k_rowstat1(const float* out, float* rmax, float* rsum) {
  int b = blockIdx.x, t = threadIdx.x;
  __shared__ float red[256];
  const float* row = out + (long)b * VP;
  float m = -3.0e38f;
  for (int v = t; v < VV; v += 256) m = fmaxf(m, row[v]);
  red[t] = m; __syncthreads();
  for (int s = 128; s > 0; s >>= 1) { if (t < s) red[t] = fmaxf(red[t], red[t + s]); __syncthreads(); }
  m = red[0]; __syncthreads();
  float sum = 0.0f;
  for (int v = t; v < VV; v += 256) sum += __expf(row[v] - m);
  red[t] = sum; __syncthreads();
  for (int s = 128; s > 0; s >>= 1) { if (t < s) red[t] += red[t + s]; __syncthreads(); }
  if (t == 0) { rmax[b] = m; rsum[b] = red[0]; }
}

// ---------- 11. final = pgen * softmax(logits), extras zeroed (in place) ----------
__global__ void __launch_bounds__(256) k_finalize(float* out, const float* rmax,
                                                  const float* rsum, const float* pgen) {
  long e = (long)blockIdx.x * 256 + threadIdx.x;
  if (e >= (long)BB * VP) return;
  int b = (int)(e / VP), v = (int)(e % VP);
  float x = 0.0f;
  if (v < VV) x = pgen[b] * __expf(out[e] - rmax[b]) / rsum[b];
  out[e] = x;
}

// ---------- 12. scatter-add copy distribution ----------
__global__ void __launch_bounds__(256) k_scatter(float* out, const int* ebev,
                                                 const float* attnw, const float* pgen) {
  int f = blockIdx.x * 256 + threadIdx.x;        // f = s*B + b, matches [S,B] layout
  if (f >= SS * BB) return;
  int s = f >> 8, b = f & 255;
  int idx = ebev[f];
  float val = (1.0f - pgen[b]) * attnw[(long)b * SS + s];
  atomicAdd(&out[(long)b * VP + idx], val);
}

// ---------- 13. per-row sum of (final + 1e-5) ----------
__global__ void __launch_bounds__(256) k_rowstat2(const float* out, float* logZ) {
  int b = blockIdx.x, t = threadIdx.x;
  __shared__ float red[256];
  const float* row = out + (long)b * VP;
  float sum = 0.0f;
  for (int v = t; v < VP; v += 256) sum += row[v] + 1.0e-5f;
  red[t] = sum; __syncthreads();
  for (int s = 128; s > 0; s >>= 1) { if (t < s) red[t] += red[t + s]; __syncthreads(); }
  if (t == 0) logZ[b] = logf(red[0]);
}

// ---------- 14. out = log(final+1e-5) - logZ (in place) ----------
__global__ void __launch_bounds__(256) k_lognorm(float* out, const float* logZ) {
  long e = (long)blockIdx.x * 256 + threadIdx.x;
  if (e >= (long)BB * VP) return;
  int b = (int)(e / VP);
  out[e] = logf(out[e] + 1.0e-5f) - logZ[b];
}

extern "C" void kernel_launch(void* const* d_in, const int* in_sizes, int n_in,
                              void* d_out, int out_size, void* d_ws, size_t ws_size,
                              hipStream_t stream) {
  const float* x      = (const float*)d_in[0];
  const float* hprev  = (const float*)d_in[1];   // [1,B,H] == [B,H]
  const float* enc    = (const float*)d_in[2];
  const int*   ebev   = (const int*)d_in[3];
  /* d_in[4] extra_zeros unused (implicit) */
  const float* mask   = (const float*)d_in[5];
  const float* Wih    = (const float*)d_in[6];
  const float* Whh    = (const float*)d_in[7];
  const float* bih    = (const float*)d_in[8];
  const float* bhh    = (const float*)d_in[9];
  const float* attnW  = (const float*)d_in[10];
  const float* attnb  = (const float*)d_in[11];
  const float* attnv  = (const float*)d_in[12];
  const float* cW     = (const float*)d_in[13];
  const float* cb     = (const float*)d_in[14];
  const float* outW   = (const float*)d_in[15];
  const float* outb   = (const float*)d_in[16];
  const float* pW     = (const float*)d_in[17];
  const float* pb     = (const float*)d_in[18];
  const float* sigW   = (const float*)d_in[19];
  const float* sigb   = (const float*)d_in[20];
  float* out = (float*)d_out;

  float* ws     = (float*)d_ws;
  float* gi     = ws;                 // 256*1536
  float* gh     = gi     + 393216;    // 256*1536
  float* cc     = gh     + 393216;    // 256*1024  [rnn | context]
  float* hpart  = cc     + 262144;    // 256*512
  float* scores = hpart  + 131072;    // 256*400
  float* attnw  = scores + 102400;    // 256*400
  float* co     = attnw  + 102400;    // 256*512
  float* pgen   = co     + 131072;    // 256
  float* rmax   = pgen   + 256;
  float* rsum   = rmax   + 256;
  float* logZ   = rsum   + 256;       // fp32 end: 1516544 floats

  __bf16* bfb    = (__bf16*)(ws + 1516544);
  __bf16* xsw    = bfb;               // 256*1024
  __bf16* hpsw   = xsw    + 262144;   // 256*512
  __bf16* Wihsw  = hpsw   + 131072;   // 1536*1024
  __bf16* Whhsw  = Wihsw  + 1572864;  // 1536*512
  __bf16* aWhsw  = Whhsw  + 786432;   // 512*512
  __bf16* aWesw  = aWhsw  + 262144;   // 512*512
  __bf16* cWsw   = aWesw  + 262144;   // 512*1024
  __bf16* rnnsw  = cWsw   + 524288;   // 256*512
  __bf16* ccsw   = rnnsw  + 131072;   // 256*1024
  __bf16* cosw   = ccsw   + 262144;   // 256*512

  // one-time fragment-major bf16 swizzles of inputs & weights
  k_swz_a<<<1024, 256, 0, stream>>>(x,     xsw,   1024, 0,   32, 262144L);
  k_swz_a<<<512,  256, 0, stream>>>(hprev, hpsw,  512,  0,   16, 131072L);
  k_swz_b<<<6144, 256, 0, stream>>>(Wih,   Wihsw, 1024, 0,   32, 1572864L);
  k_swz_b<<<3072, 256, 0, stream>>>(Whh,   Whhsw, 512,  0,   16, 786432L);
  k_swz_b<<<1024, 256, 0, stream>>>(attnW, aWhsw, 1024, 0,   16, 262144L);
  k_swz_b<<<1024, 256, 0, stream>>>(attnW, aWesw, 1024, 512, 16, 262144L);
  k_swz_b<<<2048, 256, 0, stream>>>(cW,    cWsw,  1024, 0,   32, 524288L);

  k_gru_gemm<<<dim3(16, 96), 32, 0, stream>>>(xsw, hpsw, Wihsw, Whhsw, bih, bhh, gi, gh);
  k_gates<<<256, 128, 0, stream>>>(gi, gh, hprev, sigW, sigb, cc, out);
  k_swz_a<<<512, 256, 0, stream>>>(cc, rnnsw, 1024, 0, 16, 131072L);     // rnn cols
  k_hpart<<<dim3(16, 32), 32, 0, stream>>>(rnnsw, aWhsw, hpart);
  k_energy<<<6400, 256, 0, stream>>>(enc, aWesw, attnb, attnv, hpart, scores);
  k_softmax<<<256, 256, 0, stream>>>(scores, mask, attnw, out);
  k_context<<<dim3(256, 2), 256, 0, stream>>>(enc, attnw, cc);
  k_pgen<<<256, 256, 0, stream>>>(hprev, cc, x, pW, pb, pgen);
  k_swz_a<<<1024, 256, 0, stream>>>(cc, ccsw, 1024, 0, 32, 262144L);     // [rnn|ctx]
  k_concat<<<dim3(16, 32), 32, 0, stream>>>(ccsw, cWsw, cb, co);
  k_swz_a<<<512, 256, 0, stream>>>(co, cosw, 512, 0, 16, 131072L);
  k_logits<<<dim3(4, 3142), 32, 0, stream>>>(cosw, outW, outb, out);
  k_rowstat1<<<256, 256, 0, stream>>>(out, rmax, rsum);
  k_finalize<<<50357, 256, 0, stream>>>(out, rmax, rsum, pgen);
  k_scatter<<<400, 256, 0, stream>>>(out, ebev, attnw, pgen);
  k_rowstat2<<<256, 256, 0, stream>>>(out, logZ);
  k_lognorm<<<50357, 256, 0, stream>>>(out, logZ);
}
